// GraphMemoryVQ_24902220382720
// MI455X (gfx1250) — matrix operand order repaired and checked by reading, not compiled, not run on hardware
//
#include <hip/hip_runtime.h>
#include <math.h>

typedef _Float16 f16;
typedef _Float16 v8h  __attribute__((ext_vector_type(8)));
typedef _Float16 v16h __attribute__((ext_vector_type(16)));
typedef float    v8f  __attribute__((ext_vector_type(8)));

#define GBS   0.8f
#define CCOST 0.01f

#define KDIM 512          // 2*D, fixed by the reference problem
#define NKT  16           // KDIM / 32 WMMA K-steps
#define LDST 520          // padded f16 stride per staged code row (bank-conflict avoidance)

// ---- K0: zero the loss accumulator ----
__global__ void k_zero(float* accum) {
  if (threadIdx.x == 0) accum[0] = 0.f;
}

// ---- K1: pack z (real||imag) into f16 [M][512] ----
__global__ void k_pack_z(const float* __restrict__ zr, const float* __restrict__ zi,
                         f16* __restrict__ zh, int M, int D) {
  size_t i = (size_t)blockIdx.x * blockDim.x + threadIdx.x;
  size_t total = (size_t)M * 2 * D;
  if (i >= total) return;
  int twoD = 2 * D;
  int m = (int)(i / twoD);
  int j = (int)(i % twoD);
  float v = (j < D) ? zr[(size_t)m * D + j] : zi[(size_t)m * D + (j - D)];
  zh[i] = (f16)v;
}

// ---- K2: pack codebook into f16 and compute |c_k|^2 in fp32 ----
__global__ void k_pack_codebook(const float* __restrict__ cb, f16* __restrict__ ch,
                                float* __restrict__ cnorm, int twoD) {
  __shared__ float red[256];
  int k = blockIdx.x;
  const float* row = cb + (size_t)k * twoD;
  float s = 0.f;
  for (int j = threadIdx.x; j < twoD; j += blockDim.x) {
    float v = row[j];
    ch[(size_t)k * twoD + j] = (f16)v;
    s += v * v;
  }
  red[threadIdx.x] = s;
  __syncthreads();
  for (int o = 128; o > 0; o >>= 1) {
    if ((int)threadIdx.x < o) red[threadIdx.x] += red[threadIdx.x + o];
    __syncthreads();
  }
  if (threadIdx.x == 0) cnorm[k] = red[0];
}

__device__ __forceinline__ v16h mk16(v8h lo, v8h hi) {
  v16h t;
#pragma unroll
  for (int i = 0; i < 8; ++i) { t[i] = lo[i]; t[i + 8] = hi[i]; }
  return t;
}

// ---- K3: fused distance-GEMM (WMMA f16 -> f32) + bias + argmin over K ----
// 256 threads = 8 waves; each wave owns one 16-row M tile. 32 codebook rows
// are staged per LDS tile; each wave runs 4 independent WMMA accumulator
// chains (2 K-chains x 2 N-column-tiles) for pipelining.
__global__ void __launch_bounds__(256, 1)
k_vq_main(const f16* __restrict__ zh, const f16* __restrict__ ch,
          const float* __restrict__ cnorm, const float* __restrict__ adj,
          const int* __restrict__ prev, int* __restrict__ minIdx,
          float* __restrict__ idxOut, int K) {
  __shared__ f16 ldsB[32 * LDST];
  const int lane  = threadIdx.x & 31;
  const int wave  = threadIdx.x >> 5;
  const int m0    = (blockIdx.x * 8 + wave) * 16;
  const int half  = (lane < 16) ? 0 : 1;
  const int lan16 = lane & 15;

  // Preload A fragments for 16 rows x 512 K (16-bit A layout:
  // lane holds row M=lane%16; K = half*16 + 0..15 (contiguous))
  v16h a[NKT];
  {
    const f16* zrow = zh + (size_t)(m0 + lan16) * KDIM;
#pragma unroll
    for (int kt = 0; kt < NKT; ++kt) {
      int kb = kt * 32 + half * 16;
      v8h lo = *(const v8h*)(zrow + kb);
      v8h hi = *(const v8h*)(zrow + kb + 8);
      a[kt] = mk16(lo, hi);
    }
  }

  // Per-lane adjacency/cnorm pointers, advanced by constant strides so the
  // inner loop uses immediate-offset loads (no per-iteration 64-bit adds).
  const int mb = m0 + half * 8;           // C/D layout: slot r -> M = r + half*8
  const float* pA[8];
#pragma unroll
  for (int r = 0; r < 8; ++r) pA[r] = adj + (size_t)prev[mb + r] * K + lan16;
  const float* pc = cnorm + lan16;

  float minv[8];
  int   mini[8];
#pragma unroll
  for (int r = 0; r < 8; ++r) { minv[r] = 3.4e38f; mini[r] = 0; }

  const int fcode = threadIdx.x >> 3;  // 32 staged code rows, 8 threads each
  const int fseg  = threadIdx.x & 7;   // 64-f16 (128B) segment within a row
  const int NT2 = K / 32;
  const v8f vzero = {0.f, 0.f, 0.f, 0.f, 0.f, 0.f, 0.f, 0.f};

  for (int nt = 0; nt < NT2; ++nt) {
    const int n0 = nt * 32;
    __syncthreads();  // previous tile fully consumed
    {
      const uint4* src = (const uint4*)(ch + (size_t)(n0 + fcode) * KDIM + fseg * 64);
      uint4* dst = (uint4*)(ldsB + fcode * LDST + fseg * 64);
#pragma unroll
      for (int q = 0; q < 8; ++q) dst[q] = src[q];
      if (nt + 1 < NT2)
        __builtin_prefetch(ch + (size_t)(n0 + 32 + fcode) * KDIM + fseg * 64, 0, 0);
    }
    __syncthreads();

    v8f acc0[2] = {vzero, vzero};   // columns n0 + lan16
    v8f acc1[2] = {vzero, vzero};   // columns n0 + 16 + lan16
#pragma unroll
    for (int kt = 0; kt < NKT; ++kt) {
      // B layout: lane holds column N=lane%16, K = half*16 + 0..15 (contiguous)
      const v8h* bpa = (const v8h*)(ldsB + lan16 * LDST + kt * 32 + half * 16);
      const v8h* bpb = (const v8h*)(ldsB + (16 + lan16) * LDST + kt * 32 + half * 16);
      v16h ba = mk16(bpa[0], bpa[1]);
      v16h bb = mk16(bpb[0], bpb[1]);
      acc0[kt & 1] = __builtin_amdgcn_wmma_f32_16x16x32_f16(
          false, a[kt], false, ba, (short)0, acc0[kt & 1], false, false);
      acc1[kt & 1] = __builtin_amdgcn_wmma_f32_16x16x32_f16(
          false, a[kt], false, bb, (short)0, acc1[kt & 1], false, false);
    }
    v8f s0 = acc0[0] + acc0[1];
    v8f s1 = acc1[0] + acc1[1];

    const int   na  = n0 + lan16;
    const float cn0 = pc[0];
    const float cn1 = pc[16];
#pragma unroll
    for (int r = 0; r < 8; ++r) {
      float b0 = pA[r][0];
      float b1 = pA[r][16];
      float d0 = cn0 - 2.f * s0[r] - GBS / (1.f + __expf(-b0));
      float d1 = cn1 - 2.f * s1[r] - GBS / (1.f + __expf(-b1));
      if (d0 < minv[r]) { minv[r] = d0; mini[r] = na; }
      if (d1 < minv[r]) { minv[r] = d1; mini[r] = na + 16; }
      pA[r] += 32;
    }
    pc += 32;
  }

  // reduce (val,idx) across the 16 lanes that share each M row; tie -> lower idx
#pragma unroll
  for (int r = 0; r < 8; ++r) {
#pragma unroll
    for (int off = 8; off > 0; off >>= 1) {
      float ov = __shfl_xor(minv[r], off, 16);
      int   oi = __shfl_xor(mini[r], off, 16);
      if (ov < minv[r] || (ov == minv[r] && oi < mini[r])) { minv[r] = ov; mini[r] = oi; }
    }
  }
  if (lan16 == 0) {
#pragma unroll
    for (int r = 0; r < 8; ++r) {
      minIdx[mb + r] = mini[r];
      idxOut[mb + r] = (float)mini[r];
    }
  }
}

// ---- K4: gather z_q from fp32 codebook, write outputs, accumulate loss ----
__global__ void k_gather_loss(const float* __restrict__ zr, const float* __restrict__ zi,
                              const float* __restrict__ cb, const int* __restrict__ minIdx,
                              float* __restrict__ outR, float* __restrict__ outI,
                              float* __restrict__ accum, int D) {
  __shared__ float red[256];
  int m = blockIdx.x;
  int k = minIdx[m];
  const float* crow = cb + (size_t)k * 2 * D;
  float s = 0.f;
  for (int j = threadIdx.x; j < 2 * D; j += blockDim.x) {
    float q = crow[j];
    float z;
    if (j < D) { z = zr[(size_t)m * D + j];        outR[(size_t)m * D + j] = q; }
    else       { z = zi[(size_t)m * D + (j - D)];  outI[(size_t)m * D + (j - D)] = q; }
    float dd = q - z;
    s += dd * dd;
  }
  red[threadIdx.x] = s;
  __syncthreads();
  for (int o = 128; o > 0; o >>= 1) {
    if ((int)threadIdx.x < o) red[threadIdx.x] += red[threadIdx.x + o];
    __syncthreads();
  }
  if (threadIdx.x == 0) atomicAdd(accum, red[0]);
}

// ---- K5: finalize loss = (1 + commitment_cost) * mean ----
__global__ void k_finalize(const float* __restrict__ accum, float* __restrict__ lossOut,
                           float invCount) {
  if (threadIdx.x == 0) lossOut[0] = (1.f + CCOST) * accum[0] * invCount;
}

extern "C" void kernel_launch(void* const* d_in, const int* in_sizes, int n_in,
                              void* d_out, int out_size, void* d_ws, size_t ws_size,
                              hipStream_t stream) {
  const float* zr   = (const float*)d_in[0];
  const float* zi   = (const float*)d_in[1];
  const int*   prev = (const int*)d_in[2];
  const float* cb   = (const float*)d_in[3];
  const float* adj  = (const float*)d_in[4];

  const int M    = in_sizes[2];          // B*S = 32768
  const int D    = in_sizes[0] / M;      // 256
  const int twoD = 2 * D;                // 512
  const int K    = in_sizes[3] / twoD;   // 4096

  // workspace carve-up (~38 MB)
  char* w = (char*)d_ws;
  f16*   zh     = (f16*)w;   w += (size_t)M * twoD * sizeof(f16);
  f16*   ch     = (f16*)w;   w += (size_t)K * twoD * sizeof(f16);
  float* cnorm  = (float*)w; w += (size_t)K * sizeof(float);
  int*   minIdx = (int*)w;   w += (size_t)M * sizeof(int);
  float* accum  = (float*)w;

  // output layout: [real M*D][imag M*D][loss][indices M]
  float* outR  = (float*)d_out;
  float* outI  = outR + (size_t)M * D;
  float* lossP = outI + (size_t)M * D;
  float* idxP  = lossP + 1;

  k_zero<<<1, 64, 0, stream>>>(accum);
  {
    size_t tot = (size_t)M * twoD;
    k_pack_z<<<(int)((tot + 255) / 256), 256, 0, stream>>>(zr, zi, zh, M, D);
  }
  k_pack_codebook<<<K, 256, 0, stream>>>(cb, ch, cnorm, twoD);
  k_vq_main<<<M / 128, 256, 0, stream>>>(zh, ch, cnorm, adj, prev, minIdx, idxP, K);
  k_gather_loss<<<M, 256, 0, stream>>>(zr, zi, cb, minIdx, outR, outI, accum, D);
  k_finalize<<<1, 64, 0, stream>>>(accum, lossP, 1.f / ((float)M * (float)twoD));
}
